// GeometricRelationshipEncoder_28372553957929
// MI455X (gfx1250) — matrix operand order repaired and checked by reading, not compiled, not run on hardware
//
#include <hip/hip_runtime.h>
#include <math.h>

#define BB 2
#define NN 512
#define DD 256
#define HH 128
#define NHH 8
#define HDD 32
#define SS 64
#define CHUNK 64
#define LN_EPS 1e-5f

typedef __attribute__((ext_vector_type(16))) __bf16 v16bf;
typedef __attribute__((ext_vector_type(8)))  float  v8f;

__device__ __forceinline__ unsigned short f2bf(float x) {
    unsigned u = __float_as_uint(x);
    unsigned r = u + 0x7fffu + ((u >> 16) & 1u);   // round-to-nearest-even
    return (unsigned short)(r >> 16);
}
__device__ __forceinline__ __bf16 us2bf(unsigned short u) {
    union { unsigned short u; __bf16 b; } c; c.u = u; return c.b;
}

// ---------------------------------------------------------------------------
// Kernel 0: fp32 -> bf16 weight conversion (grid-stride)
// ---------------------------------------------------------------------------
__global__ void f2bf_kernel(const float* __restrict__ src, unsigned short* __restrict__ dst, int n) {
    for (int i = blockIdx.x * blockDim.x + threadIdx.x; i < n; i += gridDim.x * blockDim.x)
        dst[i] = f2bf(src[i]);
}

__global__ void zero_kernel(float* __restrict__ p, int n) {
    for (int i = blockIdx.x * blockDim.x + threadIdx.x; i < n; i += gridDim.x * blockDim.x)
        p[i] = 0.0f;
}

// ---------------------------------------------------------------------------
// Kernel 1: fused geometric MLP + mean over j  ->  geo_ctx[b,i,:]
// One workgroup (256 threads = 8 waves) per (b,i). Dynamic LDS.
// Linear(9->128) on VALU, LN+ReLU -> bf16 LDS, Linear(128->256) via WMMA
// (ge_w2 bf16 fragments pinned in VGPRs), fused LN + running mean.
// ---------------------------------------------------------------------------
__global__ __launch_bounds__(256) void geoctx_kernel(
    const float* __restrict__ coords,
    const float* __restrict__ ge_w1, const float* __restrict__ ge_b1,
    const float* __restrict__ ge_g1, const float* __restrict__ ge_be1,
    const float* __restrict__ ge_b2, const float* __restrict__ ge_g2,
    const float* __restrict__ ge_be2,
    const unsigned short* __restrict__ w2bf,   // ge_w2 bf16, [DD][HH] row-major
    float* __restrict__ geo_ctx, unsigned short* __restrict__ geo_bf)
{
    extern __shared__ unsigned char smemRaw[];
    float* wl   = (float*)smemRaw;                 // HH*9
    float* b1l  = wl + HH * 9;                     // HH
    float* g1l  = b1l + HH;
    float* be1l = g1l + HH;
    float* b2l  = be1l + HH;                       // DD
    float* g2l  = b2l + DD;
    float* be2l = g2l + DD;
    float* outF = (float*)(smemRaw + 9216);                    // CHUNK*DD f32 (64KB)
    unsigned short* hidB = (unsigned short*)(smemRaw + 9216 + 65536); // CHUNK*HH bf16
    float* part  = (float*)(smemRaw + 9216 + 65536 + 16384);   // CHUNK*4*2
    float* stats = part + CHUNK * 8;                           // CHUNK*2

    const int b = blockIdx.x / NN;
    const int i = blockIdx.x % NN;
    const int t = threadIdx.x;

    for (int idx = t; idx < HH * 9; idx += 256) wl[idx] = ge_w1[idx];
    if (t < HH) { b1l[t] = ge_b1[t]; g1l[t] = ge_g1[t]; be1l[t] = ge_be1[t]; }
    b2l[t] = ge_b2[t]; g2l[t] = ge_g2[t]; be2l[t] = ge_be2[t];

    const float cix = coords[(b * NN + i) * 3 + 0];
    const float ciy = coords[(b * NN + i) * 3 + 1];
    const float ciz = coords[(b * NN + i) * 3 + 2];

    const int lane = t & 31, w = t >> 5;
    const int lr = lane & 15, lh = lane >> 4;

    // Pin ge_w2 B-fragments in VGPRs: wave w owns N-tiles {2w, 2w+1}, 4 K-steps.
    v16bf Bfrag[2][4];
#pragma unroll
    for (int p = 0; p < 2; ++p) {
        int nt = w * 2 + p;
#pragma unroll
        for (int ks = 0; ks < 4; ++ks) {
#pragma unroll
            for (int e = 0; e < 16; ++e) {
                int n = nt * 16 + lr;
                int k = ks * 32 + 16 * lh + e;
                Bfrag[p][ks][e] = us2bf(w2bf[n * HH + k]);
            }
        }
    }
    __syncthreads();

    float gsum = 0.0f;                 // channel t running sum over j
    const int m = t >> 2, q = t & 3;   // row-within-chunk, quarter

    for (int c = 0; c < NN / CHUNK; ++c) {
        // ---- Phase A: geo features + Linear1 + LN + ReLU -> hidB (bf16) ----
        const int j = c * CHUNK + m;
        const float* cj = &coords[(b * NN + j) * 3];
        float rx = cix - cj[0], ry = ciy - cj[1], rz = ciz - cj[2];
        float dist = sqrtf(rx * rx + ry * ry + rz * rz);
        float inv = 1.0f / fmaxf(dist, 1e-12f);
        float r0 = rx * inv, r1 = ry * inv, r2 = rz * inv;
        float f0 = dist, f6 = r1 * r2, f7 = r0 * r1, f8 = r0 * r2;

        float hv[32];
        float ps = 0.0f, ps2 = 0.0f;
#pragma unroll
        for (int cc = 0; cc < 32; ++cc) {
            int hc = q * 32 + cc;
            const float* wr = &wl[hc * 9];
            float a = b1l[hc] + f0 * wr[0] + r0 * wr[1] + r1 * wr[2] + r2 * wr[3]
                      + f6 * wr[6] + f7 * wr[7] + f8 * wr[8];
            hv[cc] = a; ps += a; ps2 += a * a;
        }
        part[(m * 4 + q) * 2 + 0] = ps;
        part[(m * 4 + q) * 2 + 1] = ps2;
        __syncthreads();
        {
            float s = 0.0f, sq = 0.0f;
#pragma unroll
            for (int qq = 0; qq < 4; ++qq) { s += part[(m * 4 + qq) * 2]; sq += part[(m * 4 + qq) * 2 + 1]; }
            float mean = s / (float)HH;
            float rstd = rsqrtf(sq / (float)HH - mean * mean + LN_EPS);
#pragma unroll
            for (int cc = 0; cc < 32; ++cc) {
                int hc = q * 32 + cc;
                float v = (hv[cc] - mean) * rstd * g1l[hc] + be1l[hc];
                hidB[m * HH + hc] = f2bf(fmaxf(v, 0.0f));
            }
        }
        __syncthreads();

        // ---- Phase B: WMMA Linear2 (64x128 @ 128x256) -> outF ----
#pragma unroll
        for (int mt = 0; mt < 4; ++mt) {
            v8f acc0 = {}; v8f acc1 = {};
#pragma unroll
            for (int ks = 0; ks < 4; ++ks) {
                v16bf a;
#pragma unroll
                for (int e = 0; e < 16; ++e) {
                    int kk = ks * 32 + 16 * lh + e;
                    a[e] = us2bf(hidB[(mt * 16 + lr) * HH + kk]);
                }
                acc0 = __builtin_amdgcn_wmma_f32_16x16x32_bf16(false, a, false, Bfrag[0][ks], (short)0, acc0, false, false);
                acc1 = __builtin_amdgcn_wmma_f32_16x16x32_bf16(false, a, false, Bfrag[1][ks], (short)0, acc1, false, false);
            }
#pragma unroll
            for (int r = 0; r < 8; ++r) {
                int row = mt * 16 + r + 8 * lh;
                outF[row * DD + (w * 2 + 0) * 16 + lr] = acc0[r];
                outF[row * DD + (w * 2 + 1) * 16 + lr] = acc1[r];
            }
        }
        __syncthreads();

        // ---- Phase C: row LN (with bias b2) + accumulate mean over j ----
        {
            float s = 0.0f, sq = 0.0f;
#pragma unroll
            for (int cc = 0; cc < 64; ++cc) {
                int col = q * 64 + cc;
                float v = outF[m * DD + col] + b2l[col];
                s += v; sq += v * v;
            }
            part[(m * 4 + q) * 2 + 0] = s;
            part[(m * 4 + q) * 2 + 1] = sq;
        }
        __syncthreads();
        if (q == 0) {
            float s = 0.0f, sq = 0.0f;
#pragma unroll
            for (int qq = 0; qq < 4; ++qq) { s += part[(m * 4 + qq) * 2]; sq += part[(m * 4 + qq) * 2 + 1]; }
            float mean = s / (float)DD;
            stats[m * 2 + 0] = mean;
            stats[m * 2 + 1] = rsqrtf(sq / (float)DD - mean * mean + LN_EPS);
        }
        __syncthreads();
        {
            float g = g2l[t], be = be2l[t], bb = b2l[t];
            for (int mm = 0; mm < CHUNK; ++mm) {
                float v = outF[mm * DD + t] + bb;
                gsum += (v - stats[mm * 2]) * stats[mm * 2 + 1] * g + be;
            }
        }
        __syncthreads();
    }

    float gm = gsum * (1.0f / (float)NN);
    geo_ctx[(b * NN + i) * DD + t] = gm;
    geo_bf[(b * NN + i) * DD + t]  = f2bf(gm);
}

// ---------------------------------------------------------------------------
// Kernel 2b: per-point segment accumulation (global f32 atomics)
// ---------------------------------------------------------------------------
__global__ __launch_bounds__(256) void seg_accum(
    const float* __restrict__ feat, const float* __restrict__ geo,
    const int* __restrict__ lab,
    float* cnt, float* sumF, float* sumG)
{
    int p = blockIdx.x;            // 0..BB*NN-1
    int b = p / NN;
    int s = lab[p];
    int t = threadIdx.x;
    atomicAdd(&sumF[(b * SS + s) * DD + t], feat[(long)p * DD + t]);
    atomicAdd(&sumG[(b * SS + s) * DD + t], geo[(long)p * DD + t]);
    if (t == 0) atomicAdd(&cnt[b * SS + s], 1.0f);
}

// ---------------------------------------------------------------------------
// Kernel 2c: aggregator MLP on segment means; comb = MLP(mf) + mg
// One workgroup (256 threads) per batch.
// ---------------------------------------------------------------------------
__global__ __launch_bounds__(256) void aggregator_kernel(
    const float* __restrict__ cnt, const float* __restrict__ sumF, const float* __restrict__ sumG,
    const float* __restrict__ w1, const float* __restrict__ b1,
    const float* __restrict__ g1, const float* __restrict__ be1,
    const float* __restrict__ w2, const float* __restrict__ b2,
    const float* __restrict__ g2, const float* __restrict__ be2,
    float* __restrict__ comb)
{
    __shared__ float hL[SS * HH];          // 32KB
    __shared__ float part[SS * 4 * 2];

    int b = blockIdx.x, t = threadIdx.x;
    int r = t >> 2, q = t & 3;
    float invd = 1.0f / fmaxf(cnt[b * SS + r], 1.0f);
    const float* mfrow = &sumF[(b * SS + r) * DD];

    // Layer 1: Linear(256->128) + LN + ReLU
    {
        float hv[32];
        float ps = 0.0f, ps2 = 0.0f;
#pragma unroll
        for (int cc = 0; cc < 32; ++cc) {
            int hc = q * 32 + cc;
            float a = b1[hc];
            for (int k = 0; k < DD; ++k) a += mfrow[k] * invd * w1[hc * DD + k];
            hv[cc] = a; ps += a; ps2 += a * a;
        }
        part[(r * 4 + q) * 2 + 0] = ps;
        part[(r * 4 + q) * 2 + 1] = ps2;
        __syncthreads();
        float s = 0.0f, sq = 0.0f;
#pragma unroll
        for (int qq = 0; qq < 4; ++qq) { s += part[(r * 4 + qq) * 2]; sq += part[(r * 4 + qq) * 2 + 1]; }
        float mean = s / (float)HH;
        float rstd = rsqrtf(sq / (float)HH - mean * mean + LN_EPS);
#pragma unroll
        for (int cc = 0; cc < 32; ++cc) {
            int hc = q * 32 + cc;
            float v = (hv[cc] - mean) * rstd * g1[hc] + be1[hc];
            hL[r * HH + hc] = fmaxf(v, 0.0f);
        }
    }
    __syncthreads();

    // Layer 2: Linear(128->256) + LN, then + mg
    {
        float ov[64];
        float ps = 0.0f, ps2 = 0.0f;
#pragma unroll
        for (int cc = 0; cc < 64; ++cc) {
            int oc = q * 64 + cc;
            float a = b2[oc];
            for (int k = 0; k < HH; ++k) a += hL[r * HH + k] * w2[oc * HH + k];
            ov[cc] = a; ps += a; ps2 += a * a;
        }
        __syncthreads();           // hL reads done before part reuse
        part[(r * 4 + q) * 2 + 0] = ps;
        part[(r * 4 + q) * 2 + 1] = ps2;
        __syncthreads();
        float s = 0.0f, sq = 0.0f;
#pragma unroll
        for (int qq = 0; qq < 4; ++qq) { s += part[(r * 4 + qq) * 2]; sq += part[(r * 4 + qq) * 2 + 1]; }
        float mean = s / (float)DD;
        float rstd = rsqrtf(sq / (float)DD - mean * mean + LN_EPS);
#pragma unroll
        for (int cc = 0; cc < 64; ++cc) {
            int oc = q * 64 + cc;
            float v = (ov[cc] - mean) * rstd * g2[oc] + be2[oc];
            comb[(b * SS + r) * DD + oc] = v + sumG[(b * SS + r) * DD + oc] * invd;
        }
    }
}

// ---------------------------------------------------------------------------
// Kernel 2d: enhanced = keep ? 0.7*feat + 0.3*comb[lab] : feat  (+ bf16 copy)
// ---------------------------------------------------------------------------
__global__ void enhance_kernel(
    const float* __restrict__ feat, const float* __restrict__ comb,
    const float* __restrict__ cnt, const int* __restrict__ lab,
    float* __restrict__ enh, unsigned short* __restrict__ enhbf)
{
    int idx = blockIdx.x * blockDim.x + threadIdx.x;
    if (idx >= BB * NN * DD) return;
    int p = idx / DD, ch = idx % DD;
    int b = p / NN;
    int s = lab[p];
    float f = feat[idx];
    float e = (cnt[b * SS + s] >= 2.0f)
                  ? 0.7f * f + 0.3f * comb[(b * SS + s) * DD + ch]
                  : f;
    enh[idx] = e;
    enhbf[idx] = f2bf(e);
}

// ---------------------------------------------------------------------------
// Generic strided bf16 WMMA GEMM: acc = A[MxK] * B[KxN]
//   A element (m,k) at A + m*sAm + k           (k contiguous)
//   B element (k,n) at Bm + k*sBk + n*sBn
// mode 0: Cf[m*sCm+n] = acc + bias
// mode 1: Cb[m*sCm+n] = bf16(acc + bias)
// mode 2: Cf[m*sCm+n] = resid[m*sRm+n] + residScale*(acc + bias)
// One wave (32 threads) per 16x16 tile; grid=(M/16, N/16).
// ---------------------------------------------------------------------------
__global__ __launch_bounds__(32) void wmma_gemm_bf16(
    const unsigned short* __restrict__ A, long sAm,
    const unsigned short* __restrict__ Bm, long sBk, long sBn,
    int Kdim, const float* __restrict__ bias, int mode,
    float* __restrict__ Cf, unsigned short* __restrict__ Cb, long sCm,
    const float* __restrict__ resid, long sRm, float residScale)
{
    int lane = threadIdx.x;
    int lr = lane & 15, lh = lane >> 4;
    int m0 = blockIdx.x * 16, n0 = blockIdx.y * 16;

    v8f acc = {};
    for (int ks = 0; ks < Kdim; ks += 32) {
        v16bf a, bf;
#pragma unroll
        for (int e = 0; e < 16; ++e) {
            int ka = ks + e + 16 * lh;
            a[e] = us2bf(A[(long)(m0 + lr) * sAm + ka]);
            int kb = ks + e + 16 * lh;
            bf[e] = us2bf(Bm[(long)kb * sBk + (long)(n0 + lr) * sBn]);
        }
        acc = __builtin_amdgcn_wmma_f32_16x16x32_bf16(false, a, false, bf, (short)0, acc, false, false);
    }
#pragma unroll
    for (int r = 0; r < 8; ++r) {
        int mrow = m0 + r + 8 * lh;
        int ncol = n0 + lr;
        float v = acc[r] + (bias ? bias[ncol] : 0.0f);
        if (mode == 0)      Cf[(long)mrow * sCm + ncol] = v;
        else if (mode == 1) Cb[(long)mrow * sCm + ncol] = f2bf(v);
        else                Cf[(long)mrow * sCm + ncol] =
                                resid[(long)mrow * sRm + ncol] + residScale * v;
    }
}

// ---------------------------------------------------------------------------
// Softmax over rows of 512, scaled by 1/sqrt(hd); writes bf16 probabilities.
// ---------------------------------------------------------------------------
__global__ __launch_bounds__(256) void softmax_kernel(
    const float* __restrict__ scores, unsigned short* __restrict__ P, float scale)
{
    __shared__ float red[256];
    int row = blockIdx.x;
    const float* s = scores + (long)row * NN;
    int t = threadIdx.x;
    float a0 = s[t], a1 = s[t + 256];
    red[t] = fmaxf(a0, a1);
    __syncthreads();
    for (int o = 128; o > 0; o >>= 1) {
        if (t < o) red[t] = fmaxf(red[t], red[t + o]);
        __syncthreads();
    }
    float mx = red[0];
    __syncthreads();
    float e0 = __expf((a0 - mx) * scale), e1 = __expf((a1 - mx) * scale);
    red[t] = e0 + e1;
    __syncthreads();
    for (int o = 128; o > 0; o >>= 1) {
        if (t < o) red[t] += red[t + o];
        __syncthreads();
    }
    float invs = 1.0f / red[0];
    P[(long)row * NN + t]       = f2bf(e0 * invs);
    P[(long)row * NN + t + 256] = f2bf(e1 * invs);
}

// ---------------------------------------------------------------------------
extern "C" void kernel_launch(void* const* d_in, const int* in_sizes, int n_in,
                              void* d_out, int out_size, void* d_ws, size_t ws_size,
                              hipStream_t stream)
{
    (void)in_sizes; (void)n_in; (void)out_size; (void)ws_size;

    const float* coords   = (const float*)d_in[0];
    const float* features = (const float*)d_in[1];
    const int*   labels   = (const int*)  d_in[2];
    const float* ge_w1  = (const float*)d_in[3];
    const float* ge_b1  = (const float*)d_in[4];
    const float* ge_g1  = (const float*)d_in[5];
    const float* ge_be1 = (const float*)d_in[6];
    const float* ge_w2  = (const float*)d_in[7];
    const float* ge_b2  = (const float*)d_in[8];
    const float* ge_g2  = (const float*)d_in[9];
    const float* ge_be2 = (const float*)d_in[10];
    const float* ag_w1  = (const float*)d_in[11];
    const float* ag_b1  = (const float*)d_in[12];
    const float* ag_g1  = (const float*)d_in[13];
    const float* ag_be1 = (const float*)d_in[14];
    const float* ag_w2  = (const float*)d_in[15];
    const float* ag_b2  = (const float*)d_in[16];
    const float* ag_g2  = (const float*)d_in[17];
    const float* ag_be2 = (const float*)d_in[18];
    const float* wq = (const float*)d_in[19];
    const float* bq = (const float*)d_in[20];
    const float* wk = (const float*)d_in[21];
    const float* bk = (const float*)d_in[22];
    const float* wv = (const float*)d_in[23];
    const float* bv = (const float*)d_in[24];
    const float* wo = (const float*)d_in[25];
    const float* bo = (const float*)d_in[26];

    unsigned char* ws = (unsigned char*)d_ws;
    size_t off = 0;
    auto nxt = [&](size_t bytes) -> unsigned char* {
        unsigned char* p = ws + off;
        off += (bytes + 255) & ~(size_t)255;
        return p;
    };
    unsigned short* w2ge_bf = (unsigned short*)nxt((size_t)DD * HH * 2);
    unsigned short* wq_bf   = (unsigned short*)nxt((size_t)DD * DD * 2);
    unsigned short* wk_bf   = (unsigned short*)nxt((size_t)DD * DD * 2);
    unsigned short* wv_bf   = (unsigned short*)nxt((size_t)DD * DD * 2);
    unsigned short* wo_bf   = (unsigned short*)nxt((size_t)DD * DD * 2);
    float*          geo_ctx = (float*)         nxt((size_t)BB * NN * DD * 4);
    unsigned short* geo_bf  = (unsigned short*)nxt((size_t)BB * NN * DD * 2);
    float*          enh     = (float*)         nxt((size_t)BB * NN * DD * 4);
    unsigned short* enh_bf  = (unsigned short*)nxt((size_t)BB * NN * DD * 2);
    float*          cnt     = (float*)         nxt((size_t)BB * SS * 4);
    float*          sumF    = (float*)         nxt((size_t)BB * SS * DD * 4);
    float*          sumG    = (float*)         nxt((size_t)BB * SS * DD * 4);
    float*          comb    = (float*)         nxt((size_t)BB * SS * DD * 4);
    unsigned short* Qbf     = (unsigned short*)nxt((size_t)BB * NN * DD * 2);
    unsigned short* Kbf     = (unsigned short*)nxt((size_t)BB * NN * DD * 2);
    unsigned short* Vbf     = (unsigned short*)nxt((size_t)BB * NN * DD * 2);
    float*          scoresF = (float*)         nxt((size_t)NN * NN * 4);
    unsigned short* Pbf     = (unsigned short*)nxt((size_t)NN * NN * 2);
    unsigned short* Obf     = (unsigned short*)nxt((size_t)BB * NN * DD * 2);

    // 0. weights -> bf16
    f2bf_kernel<<<(DD * HH + 255) / 256, 256, 0, stream>>>(ge_w2, w2ge_bf, DD * HH);
    f2bf_kernel<<<(DD * DD + 255) / 256, 256, 0, stream>>>(wq, wq_bf, DD * DD);
    f2bf_kernel<<<(DD * DD + 255) / 256, 256, 0, stream>>>(wk, wk_bf, DD * DD);
    f2bf_kernel<<<(DD * DD + 255) / 256, 256, 0, stream>>>(wv, wv_bf, DD * DD);
    f2bf_kernel<<<(DD * DD + 255) / 256, 256, 0, stream>>>(wo, wo_bf, DD * DD);

    // 1. fused geo MLP + mean -> geo_ctx (the 95%-of-FLOPs kernel)
    geoctx_kernel<<<BB * NN, 256, 93696, stream>>>(
        coords, ge_w1, ge_b1, ge_g1, ge_be1, ge_b2, ge_g2, ge_be2,
        w2ge_bf, geo_ctx, geo_bf);

    // 2. superpoint segment means + aggregator MLP + enhanced
    zero_kernel<<<(BB * SS + 255) / 256, 256, 0, stream>>>(cnt, BB * SS);
    zero_kernel<<<(BB * SS * DD + 255) / 256, 256, 0, stream>>>(sumF, BB * SS * DD);
    zero_kernel<<<(BB * SS * DD + 255) / 256, 256, 0, stream>>>(sumG, BB * SS * DD);
    seg_accum<<<BB * NN, 256, 0, stream>>>(features, geo_ctx, labels, cnt, sumF, sumG);
    aggregator_kernel<<<BB, 256, 0, stream>>>(cnt, sumF, sumG,
        ag_w1, ag_b1, ag_g1, ag_be1, ag_w2, ag_b2, ag_g2, ag_be2, comb);
    enhance_kernel<<<(BB * NN * DD + 255) / 256, 256, 0, stream>>>(
        features, comb, cnt, labels, enh, enh_bf);

    // 3. Q/K/V projections (WMMA): rows = B*N = 1024, K = N = 256
    dim3 gp(BB * NN / 16, DD / 16);
    wmma_gemm_bf16<<<gp, 32, 0, stream>>>(enh_bf, DD, wq_bf, 1, DD, DD, bq, 1,
                                          nullptr, Qbf, DD, nullptr, 0, 0.0f);
    wmma_gemm_bf16<<<gp, 32, 0, stream>>>(geo_bf, DD, wk_bf, 1, DD, DD, bk, 1,
                                          nullptr, Kbf, DD, nullptr, 0, 0.0f);
    wmma_gemm_bf16<<<gp, 32, 0, stream>>>(geo_bf, DD, wv_bf, 1, DD, DD, bv, 1,
                                          nullptr, Vbf, DD, nullptr, 0, 0.0f);

    // 4. attention per (b,h): QK^T (WMMA) -> softmax -> PV (WMMA)
    const float scale = 0.17677669529663687f; // 1/sqrt(32)
    for (int b = 0; b < BB; ++b) {
        for (int h = 0; h < NHH; ++h) {
            long o = (long)b * NN * DD + (long)h * HDD;
            dim3 gs(NN / 16, NN / 16);
            wmma_gemm_bf16<<<gs, 32, 0, stream>>>(Qbf + o, DD, Kbf + o, 1, DD, HDD,
                                                  nullptr, 0, scoresF, nullptr, NN,
                                                  nullptr, 0, 0.0f);
            softmax_kernel<<<NN, 256, 0, stream>>>(scoresF, Pbf, scale);
            dim3 gpv(NN / 16, HDD / 16);
            wmma_gemm_bf16<<<gpv, 32, 0, stream>>>(Pbf, NN, Vbf + o, DD, 1, NN,
                                                   nullptr, 1, nullptr, Obf + o, DD,
                                                   nullptr, 0, 0.0f);
        }
    }

    // 5. output projection + residual: out = enhanced + 0.5*(O @ wo^T + bo)
    dim3 gf(BB * NN / 16, DD / 16);
    wmma_gemm_bf16<<<gf, 32, 0, stream>>>(Obf, DD, wo_bf, 1, DD, DD, bo, 2,
                                          (float*)d_out, nullptr, DD, enh, DD, 0.5f);
}
